// Attention_6356551598815
// MI455X (gfx1250) — matrix operand order repaired and checked by reading, not compiled, not run on hardware
//
#include <hip/hip_runtime.h>
#include <stdint.h>

typedef __attribute__((ext_vector_type(16))) __bf16 v16bf;
typedef __attribute__((ext_vector_type(8)))  __bf16 v8bf;
typedef __attribute__((ext_vector_type(4)))  __bf16 v4bf;
typedef __attribute__((ext_vector_type(8)))  float  v8f;
typedef __attribute__((ext_vector_type(4)))  float  v4f;
typedef __attribute__((ext_vector_type(4)))  unsigned int v4u;
typedef __attribute__((ext_vector_type(8)))  int v8i;
typedef __attribute__((ext_vector_type(4)))  int v4i;

#define B_  32
#define T_  2048
#define D_  1024
#define U_  1024
#define TM  128      // t-rows per block
#define LDA 1032     // bf16 elements per LDS A row (1024 + 8 pad -> stride%64dw==4, conflict free)
#define BROW 40      // bf16 per LDS B row (32 data + 8 pad -> 80B stride, conflict free)

#if __has_builtin(__builtin_amdgcn_tensor_load_to_lds)
#define HAVE_TDM 1
#else
#define HAVE_TDM 0
#endif

// ---------------- kernel 1: W1 fp32 -> bf16, B-fragment-friendly packing -------------
// W1p[(kb*U + u)*32 + j] = bf16(W1[(kb*32 + j)*U + u]) ; per-u 32-k chunk = 64B contiguous
__global__ void k_pack_w1(const float* __restrict__ W1, unsigned short* __restrict__ W1p) {
  int i = blockIdx.x * 256 + threadIdx.x;          // i = k*U + u, i < 1M
  int k = i >> 10, u = i & (U_ - 1);
  int kb = k >> 5, j = k & 31;
  __bf16 v = (__bf16)W1[i];
  W1p[((size_t)(kb * U_ + u) << 5) + j] = __builtin_bit_cast(unsigned short, v);
}

// ---------------- kernel 2: q2[b,u] = W1_b[u] + W2_b[u] + query[b,:] @ W2[:,u] --------
__global__ void k_q2(const float* __restrict__ query, const float* __restrict__ W2k,
                     const float* __restrict__ W1b, const float* __restrict__ W2b,
                     float* __restrict__ q2) {
  int u = blockIdx.x * 256 + threadIdx.x;
  int b = blockIdx.y;
  float acc = W1b[u] + W2b[u];
  const float* q = query + (size_t)b * D_;
#pragma unroll 4
  for (int d = 0; d < D_; ++d) acc = fmaf(q[d], W2k[(size_t)d * U_ + u], acc);
  q2[(size_t)b * U_ + u] = acc;
}

// ---------------- kernel 3: fused GEMM(bf16 WMMA) + tanh + V-dot -> score[b,t] -------
struct SmemB {
  __bf16 tileA[TM * LDA];        // 264,192 B
  __bf16 Bbuf[2][128 * BROW];    //  20,480 B (double-buffered W1 K-chunk via TDM)
  float  q2s[U_];                //   4,096 B
  float  vks[U_];                //   4,096 B
  float  score[TM];              //     512 B   -> total 293,376 B < 320 KB
};

#if HAVE_TDM
// Build Tensor-DMA descriptor (ISA 08 §8): 2D tile, 2-byte elems, tile 32x128,
// row stride 32 elems, LDS pad 16B after every 64B (interval code 3, amount code 3).
__device__ __forceinline__ void tdm_issue(const unsigned short* gp, unsigned lds_off) {
  unsigned long long ga = (unsigned long long)(uintptr_t)gp;
  v4u g0;
  g0[0] = 1u;                                              // count=1 (valid), user mode
  g0[1] = lds_off;                                         // lds_addr
  g0[2] = (unsigned)ga;                                    // global_addr[31:0]
  g0[3] = (unsigned)((ga >> 32) & 0x01FFFFFFu) | (2u << 30); // global_addr[56:32] | type=2
  v8i g1;
  g1[0] = (int)((1u << 16) | (1u << 20) | (3u << 22) | (3u << 25)); // ds=2B,pad_en,ivl=16dw,amt=4dw
  g1[1] = (int)(32u << 16);    // tensor_dim0 = 32
  g1[2] = (int)(128u << 16);   // tensor_dim1 = 128
  g1[3] = (int)(32u << 16);    // tile_dim0 = 32
  g1[4] = 128;                 // tile_dim1 = 128, tile_dim2 = 0
  g1[5] = 32;                  // tensor_dim0_stride = 32
  g1[6] = 0;                   // stride hi / dim1_stride lo
  g1[7] = 0;
  v4i g2 = {0, 0, 0, 0};
  v4i g3 = {0, 0, 0, 0};
#if defined(__clang_major__) && __clang_major__ >= 23
  v8i g4 = {0, 0, 0, 0, 0, 0, 0, 0};   // clang-23 / therock-10.0: 6-arg form
  __builtin_amdgcn_tensor_load_to_lds(g0, g1, g2, g3, g4, 0);
#else
  __builtin_amdgcn_tensor_load_to_lds(g0, g1, g2, g3, 0);  // ROCm 7.2 / clang-22: 5-arg
#endif
}
#endif

__global__ void __launch_bounds__(256)
k_score(const float* __restrict__ values, const unsigned short* __restrict__ W1p_,
        const float* __restrict__ q2, const float* __restrict__ Vk,
        const float* __restrict__ Vb, float* __restrict__ scoreg) {
  __shared__ SmemB sm;
  const int tid = threadIdx.x;
  const int b = blockIdx.y;
  const int t0 = blockIdx.x * TM;
  const int w = tid >> 5, lane = tid & 31;
  const int ln = lane & 15, half = lane >> 4;
  const __bf16* W1p = (const __bf16*)W1p_;

  // preload q2 / V_k, init score
  for (int i = tid; i < U_; i += 256) {
    sm.q2s[i] = q2[(size_t)b * U_ + i];
    sm.vks[i] = Vk[i];
  }
  if (tid < TM) sm.score[tid] = 0.f;

  // A tile: values fp32 global -> bf16 LDS (stride LDA)
  const float* vbase = values + ((size_t)b * T_ + t0) * D_;
  for (int it = tid; it < TM * (D_ / 4); it += 256) {
    int r = it >> 8;
    int c = (it & 255) << 2;
    v4f f = *(const v4f*)(vbase + (size_t)r * D_ + c);
    v4bf h;
    h[0] = (__bf16)f[0]; h[1] = (__bf16)f[1]; h[2] = (__bf16)f[2]; h[3] = (__bf16)f[3];
    *(v4bf*)&sm.tileA[r * LDA + c] = h;
  }

#if HAVE_TDM
  unsigned bboff[2];
  bboff[0] = (unsigned)(unsigned long long)(uintptr_t)&sm.Bbuf[0][0];
  bboff[1] = (unsigned)(unsigned long long)(uintptr_t)&sm.Bbuf[1][0];
  if (tid < 32) tdm_issue(W1p_, bboff[0]);   // prologue: (uc=0, kb=0) -> Bbuf[0]
#endif

  const __bf16* arow = &sm.tileA[(w * 16 + ln) * LDA];
  float acc[8] = {0.f, 0.f, 0.f, 0.f, 0.f, 0.f, 0.f, 0.f};

  for (int uc = 0; uc < 8; ++uc) {
    v8f C[8] = {};
    for (int kb = 0; kb < 32; ++kb) {
#if HAVE_TDM
      if (tid < 32) {                       // wave 0 drives the TDM pipeline
        int nk = kb + 1, nu = uc;
        if (nk == 32) { nk = 0; nu++; }
        if (nu < 8) {
          tdm_issue(W1p_ + ((size_t)(nk * U_ + nu * 128) << 5), bboff[nk & 1]);
          __builtin_amdgcn_s_wait_tensorcnt(1);   // current chunk landed (in-order)
        } else {
          __builtin_amdgcn_s_wait_tensorcnt(0);
        }
      }
      __syncthreads();                      // Bbuf[kb&1] (and tileA on first iter) ready
      const __bf16* bb = &sm.Bbuf[kb & 1][0];
#else
      __syncthreads();                      // prior readers of Bbuf[0] done
      {
        int r = tid >> 1, hh = tid & 1;     // 128 rows x 2 halves = 256 threads
        *(v8bf*)&sm.Bbuf[0][r * BROW + hh * 16] =
            *(const v8bf*)(W1p + ((size_t)(kb * U_ + uc * 128 + r) << 5) + hh * 16);
      }
      __syncthreads();
      const __bf16* bb = &sm.Bbuf[0][0];
#endif
      // A fragment (16x32 bf16): lane ln = row, VGPR0-3: K=half*8+0..7, VGPR4-7: +16
      const __bf16* ap = arow + kb * 32 + half * 8;
      v8bf a0 = *(const v8bf*)ap;
      v8bf a1 = *(const v8bf*)(ap + 16);
      v16bf A = __builtin_shufflevector(a0, a1, 0, 1, 2, 3, 4, 5, 6, 7,
                                        8, 9, 10, 11, 12, 13, 14, 15);
#pragma unroll
      for (int s = 0; s < 8; ++s) {
        const __bf16* bp = bb + (s * 16 + ln) * BROW + half * 8;
        v8bf b0 = *(const v8bf*)bp;
        v8bf b1 = *(const v8bf*)(bp + 16);
        v16bf Bf = __builtin_shufflevector(b0, b1, 0, 1, 2, 3, 4, 5, 6, 7,
                                           8, 9, 10, 11, 12, 13, 14, 15);
        C[s] = __builtin_amdgcn_wmma_f32_16x16x32_bf16(
            false, A, false, Bf, (short)0, C[s], false, false);
      }
      __syncthreads();                      // done reading Bbuf[kb&1]
    }
    // epilogue: h = C + q2 ; score-partials += tanh(h) * V_k[u]
#pragma unroll
    for (int s = 0; s < 8; ++s) {
      int u = uc * 128 + s * 16 + ln;       // C: lane%16 = column u
      float qv = sm.q2s[u], kv = sm.vks[u];
#pragma unroll
      for (int v = 0; v < 8; ++v)
        acc[v] += tanhf(C[s][v] + qv) * kv; // C: VGPR v -> row v + 8*half
    }
  }
#pragma unroll
  for (int v = 0; v < 8; ++v)
    atomicAdd(&sm.score[w * 16 + half * 8 + v], acc[v]);
  __syncthreads();
  if (tid < TM) scoreg[(size_t)b * T_ + t0 + tid] = sm.score[tid] + Vb[0];
}

// ---------------- kernel 4: softmax over T + context gather at t = T-1 ---------------
__global__ void k_softmax_ctx(const float* __restrict__ scoreg,
                              const float* __restrict__ values,
                              float* __restrict__ out) {
  __shared__ float red[256];
  int b = blockIdx.x, tid = threadIdx.x;
  const float* srow = scoreg + (size_t)b * T_;
  float m = -3.4e38f;
  for (int t = tid; t < T_; t += 256) m = fmaxf(m, srow[t]);
  red[tid] = m; __syncthreads();
  for (int s = 128; s > 0; s >>= 1) {
    if (tid < s) red[tid] = fmaxf(red[tid], red[tid + s]);
    __syncthreads();
  }
  float mx = red[0];
  __syncthreads();
  float sum = 0.f;
  for (int t = tid; t < T_; t += 256) sum += expf(srow[t] - mx);
  red[tid] = sum; __syncthreads();
  for (int s = 128; s > 0; s >>= 1) {
    if (tid < s) red[tid] += red[tid + s];
    __syncthreads();
  }
  float inv = 1.f / red[0];
  float* attn = out + B_ * D_;                       // outputs: [context | attn]
  for (int t = tid; t < T_; t += 256) attn[(size_t)b * T_ + t] = expf(srow[t] - mx) * inv;
  float wl = expf(srow[T_ - 1] - mx) * inv;
  const float* vlast = values + ((size_t)b * T_ + (T_ - 1)) * D_;
  for (int d = tid; d < D_; d += 256) out[(size_t)b * D_ + d] = wl * vlast[d];
}

// ---------------- launcher -----------------------------------------------------------
extern "C" void kernel_launch(void* const* d_in, const int* in_sizes, int n_in,
                              void* d_out, int out_size, void* d_ws, size_t ws_size,
                              hipStream_t stream) {
  (void)in_sizes; (void)n_in; (void)out_size; (void)ws_size;
  const float* query  = (const float*)d_in[0];
  const float* values = (const float*)d_in[1];
  const float* W1k    = (const float*)d_in[2];
  const float* W1b    = (const float*)d_in[3];
  const float* W2k    = (const float*)d_in[4];
  const float* W2b    = (const float*)d_in[5];
  const float* Vk     = (const float*)d_in[6];
  const float* Vb     = (const float*)d_in[7];

  char* ws = (char*)d_ws;                 // ws: [0,2MB) W1 bf16 | 128KB q2 | 256KB score
  unsigned short* W1p = (unsigned short*)ws;
  float* q2     = (float*)(ws + (size_t)2 * 1024 * 1024);
  float* scoreg = (float*)(ws + (size_t)2 * 1024 * 1024 + 128 * 1024);
  float* out = (float*)d_out;

  k_pack_w1<<<dim3((D_ * U_) / 256), dim3(256), 0, stream>>>(W1k, W1p);
  k_q2<<<dim3(U_ / 256, B_), dim3(256), 0, stream>>>(query, W2k, W1b, W2b, q2);
  k_score<<<dim3(T_ / TM, B_), dim3(256), 0, stream>>>(values, W1p, q2, Vk, Vb, scoreg);
  k_softmax_ctx<<<dim3(B_), dim3(256), 0, stream>>>(scoreg, values, out);
}
